// Finder_66640712565346
// MI455X (gfx1250) — 8-bit matrix operand order repaired and checked by reading, compiled, not run on hardware
//
#include <hip/hip_runtime.h>

typedef int v8i __attribute__((ext_vector_type(8)));

#define NB   64
#define LEXP 8192
#define NF   100
#define FT   112
#define NFT  7
#define PCH  256
#define NCHUNK 33
#define ROWS (PCH + 31)
#define PITCH 116
#define WIN  368

__global__ void finder_init_ws(unsigned long long* ws) {
    int t = threadIdx.x;
    if (t < NB) ws[t] = 0ull;
}

__global__ __launch_bounds__(256) void finder_main(const int* __restrict__ expr,
                                                   const int* __restrict__ sub,
                                                   unsigned long long* __restrict__ ws) {
    __shared__ unsigned char eLds[ROWS * PITCH];
    __shared__ int tokLds[WIN];
    __shared__ int subLds[FT];
    __shared__ unsigned long long blockKey;

    const int chunk = blockIdx.x;
    const int b     = blockIdx.y;
    const int p0    = chunk * PCH;
    const int pcur  = min(PCH, (LEXP + 1) - p0);

    const int tid  = threadIdx.x;
    const int lane = tid & 31;
    const int wave = tid >> 5;
    const int nrow = lane & 15;
    const int half = lane >> 4;

    if (tid == 0) blockKey = 0ull;

    const int* exprRow = expr + (size_t)b * LEXP;
#pragma unroll
    for (int j0 = 0; j0 < WIN; j0 += 256) {
        const int j = j0 + tid;
        if (j < WIN) {
            const int ci = min(p0 + j, LEXP - 1);
            const unsigned ldsAddr = (unsigned)(size_t)&tokLds[j];
            asm volatile("global_load_async_to_lds_b32 %0, %1, %2"
                         :: "v"(ldsAddr), "v"(ci * 4), "s"(exprRow)
                         : "memory");
        }
    }
    asm volatile("s_wait_asynccnt 0x0" ::: "memory");
#pragma unroll
    for (int j0 = 0; j0 < WIN; j0 += 256) {
        const int j = j0 + tid;
        if (j < WIN && (p0 + j) >= LEXP) tokLds[j] = -1;
    }
    if (tid < FT) {
        const int t = (tid < NF) ? sub[b * NF + tid] : 0;
        subLds[tid] = (t == 0) ? -1 : t;
    }
    __syncthreads();

    for (int tf = 0; tf < NFT; ++tf) {
        const int f0 = tf * 16;
        const int f  = f0 + nrow;

        const int tb = subLds[f];
        v8i bm;
#pragma unroll
        for (int v = 0; v < 8; ++v) {
            const int kb = ((v & 1) << 2) | (((v >> 1) & 1) << 4) | ((v >> 2) << 5) | (half << 3);
            const int d  = tb - kb;
            bm[v] = (d >= 0 && d < 4) ? (int)(1u << (d * 8)) : 0;
        }

        for (int s = wave; s <= 16; s += 8) {
            const int off = 16 * (tf + s);

            const int ta = tokLds[off + nrow];
            v8i a;
#pragma unroll
            for (int v = 0; v < 8; ++v) {
                const int kb = ((v & 1) << 2) | (((v >> 1) & 1) << 4) | ((v >> 2) << 5) | (half << 3);
                const int d  = ta - kb;
                a[v] = (d >= 0 && d < 4) ? (int)(1u << (d * 8)) : 0;
            }

            v8i c = {};
            v8i dm = __builtin_amdgcn_wmma_i32_16x16x64_iu8(false, a, false, bm, c, false, false);

#pragma unroll
            for (int r = 0; r < 8; ++r) {
                const int M   = r + (half << 3);
                const int row = off + M - f + 15;
                eLds[row * PITCH + f] = (unsigned char)dm[r];
            }
        }
    }

    __syncthreads();

    for (int idx = tid; idx < pcur; idx += 256) {
        const unsigned* rowp = (const unsigned*)&eLds[(idx + 15) * PITCH];
        int score = 0;
#pragma unroll
        for (int w = 0; w < FT / 4; ++w) score += __popc(rowp[w]);
        const unsigned p = (unsigned)(p0 + idx);
        const unsigned long long key =
            ((unsigned long long)(unsigned)score << 32) | (unsigned)(~p);
        atomicMax(&blockKey, key);
    }

    __syncthreads();
    if (tid == 0) atomicMax(&ws[b], blockKey);
}

__global__ void finder_final(const unsigned long long* __restrict__ ws,
                             float* __restrict__ out) {
    int t = threadIdx.x;
    if (t < NB) {
        const unsigned long long key = ws[t];
        const unsigned pos = ~(unsigned)(key & 0xFFFFFFFFull);
        const unsigned sc  = (unsigned)(key >> 32);
        out[t]      = (float)pos;
        out[NB + t] = (float)sc;
    }
}

extern "C" void kernel_launch(void* const* d_in, const int* in_sizes, int n_in,
                              void* d_out, int out_size, void* d_ws, size_t ws_size,
                              hipStream_t stream) {
    const int* expr = (const int*)d_in[0];
    const int* sub  = (const int*)d_in[1];
    float* out = (float*)d_out;
    unsigned long long* ws = (unsigned long long*)d_ws;

    finder_init_ws<<<1, 64, 0, stream>>>(ws);
    finder_main<<<dim3(NCHUNK, NB), 256, 0, stream>>>(expr, sub, ws);
    finder_final<<<1, 64, 0, stream>>>(ws, out);
}
